// GraphAttentionGenerator_7438883357373
// MI455X (gfx1250) — matrix operand order repaired and checked by reading, not compiled, not run on hardware
//
#include <hip/hip_runtime.h>
#include <hip/hip_bf16.h>
#include <math.h>

#define DEVFN static __device__ __forceinline__

typedef __attribute__((ext_vector_type(16))) __bf16 v16bf;
typedef __attribute__((ext_vector_type(2)))  __bf16 v2bf;
typedef __attribute__((ext_vector_type(8)))  float  v8f;
typedef __attribute__((ext_vector_type(4)))  unsigned int u32x4;
typedef __attribute__((ext_vector_type(4)))  float f32x4;
typedef __attribute__((ext_vector_type(4)))  unsigned int v4u;
typedef __attribute__((ext_vector_type(8)))  int v8i;
typedef __attribute__((ext_vector_type(4)))  int v4i;

union FragB16 { v16bf v; u32x4 q[2]; };

constexpr int B_        = 32;
constexpr int N_        = 16384;
constexpr int D_        = 128;         // QD == KD == H == 128
constexpr int THREADS_A = 128;         // 4 waves (wave32)
constexpr int NODES_IT  = 64;          // 4 waves x 16 rows per iteration
constexpr int ITERS     = 8;
constexpr int CHUNK     = NODES_IT * ITERS;   // 512 nodes per block
constexpr int CHUNKS    = N_ / CHUNK;         // 32
constexpr int PSTRIDE   = 132;                // floats per partial: m, l, acc[128], pad
constexpr float NEG_SLOPE = 0.2f;

#if defined(__AMDGCN__) && __has_builtin(__builtin_amdgcn_tensor_load_to_lds)
#define USE_TDM 1
#else
#define USE_TDM 0
#endif

DEVFN unsigned short f2bf(float f) {          // fp32 -> bf16 (round to nearest)
  return (unsigned short)((__float_as_uint(f) + 0x8000u) >> 16);
}

// pack two fp32 into two bf16 (one dword); prefer hw packed convert
#if defined(__AMDGCN__) && __has_builtin(__builtin_amdgcn_cvt_pk_bf16_f32)
DEVFN unsigned pk2(float a, float b) {
  union { v2bf v; unsigned u; } r;
  r.v = __builtin_amdgcn_cvt_pk_bf16_f32(a, b);
  return r.u;
}
#else
DEVFN unsigned pk2(float a, float b) {
  unsigned ua = __float_as_uint(a) + 0x8000u;
  unsigned ub = __float_as_uint(b) + 0x8000u;
  return (ua >> 16) | (ub & 0xFFFF0000u);
}
#endif

#if USE_TDM
// Issue one TDM 2D tile DMA: rows x 128 fp32, row stride 128 floats, HBM -> LDS.
// D# layout per CDNA5 ISA ch.8. Tracked by TENSORcnt.
DEVFN void tdm_issue_tile(unsigned lds_off, const float* gsrc, int rows) {
  unsigned long long ga = (unsigned long long)(uintptr_t)gsrc;
  v4u g0;
  g0.x = 1u;                                   // count = 1 valid descriptor
  g0.y = lds_off;                              // lds_addr (bytes)
  g0.z = (unsigned)ga;                         // global_addr[31:0]
  g0.w = (unsigned)((ga >> 32) & 0x1FFFFFFu)   // global_addr[56:32]
       | 0x80000000u;                          // type = 2 ("image")
  v8i g1;
  g1[0] = 0x20000;                             // data_size=2 (4B), wg_mask=0
  g1[1] = (int)(128u << 16);                   // tensor_dim0 = 128 (lo16)
  g1[2] = (int)((unsigned)rows << 16);         // td0 hi=0 | tensor_dim1 lo=rows
  g1[3] = (int)(128u << 16);                   // td1 hi=0 | tile_dim0 = 128
  g1[4] = rows;                                // tile_dim1 = rows, tile_dim2 = 0
  g1[5] = 128;                                 // tensor_dim0_stride = 128 (lo32)
  g1[6] = 0;
  g1[7] = 0;
  v4i z4 = (v4i)0;
#if defined(__clang_major__) && __clang_major__ >= 23
  v8i z8 = (v8i)0;
  __builtin_amdgcn_tensor_load_to_lds(g0, g1, z4, z4, z8, 0);
#else
  __builtin_amdgcn_tensor_load_to_lds(g0, g1, z4, z4, 0);
#endif
}
DEVFN void tdm_wait_all() {
#if __has_builtin(__builtin_amdgcn_s_wait_tensorcnt)
  __builtin_amdgcn_s_wait_tensorcnt(0);
#endif
}
#endif

// ---------------------------------------------------------------------------
// Kernel 0: h_query[b,h] = query[b,:] . Wq[h,:]   (fp32 exact, tiny)
// ---------------------------------------------------------------------------
__global__ __launch_bounds__(128)
void ga_hquery(const float* __restrict__ query, const float* __restrict__ Wq,
               float* __restrict__ hq) {
  int b = blockIdx.x, t = threadIdx.x;
  const f32x4* q = (const f32x4*)(query + b * D_);
  const f32x4* w = (const f32x4*)(Wq + t * D_);
  float s = 0.0f;
  #pragma unroll
  for (int i = 0; i < 32; ++i) {
    f32x4 a = q[i], c = w[i];
    s += a.x * c.x + a.y * c.y + a.z * c.z + a.w * c.w;
  }
  hq[b * D_ + t] = s;
}

// ---------------------------------------------------------------------------
// Kernel A: single pass over key, TDM double-buffered tiles.
// ---------------------------------------------------------------------------
__global__ __launch_bounds__(THREADS_A)
void ga_main(const float* __restrict__ key,
             const float* __restrict__ hquery,
             const float* __restrict__ Wk,
             const float* __restrict__ Wa,
             float* __restrict__ elog,          // d_out e-region: raw logits
             float* __restrict__ part) {        // [B*CHUNKS*PSTRIDE]
  __shared__ unsigned short lds_wk[4 * 8 * 32 * 16];   // WMMA B frags, 32 KB
  __shared__ float lds_keyf[2][NODES_IT * D_];         // fp32 tiles, 2 x 32 KB
  __shared__ float lds_hq[D_];
  __shared__ float lds_wa[D_];
  __shared__ float lds_lg[NODES_IT];
  __shared__ float lds_w[NODES_IT];
  __shared__ float lds_red[2];

  const int t  = threadIdx.x;
  const int ln = t & 31;
  const int wv = t >> 5;
  const int b  = blockIdx.y;
  const int ch = blockIdx.x;

  // ---- prologue: swizzle Wk (B = Wk^T) into per-lane bf16 fragments ----
  #pragma unroll
  for (int e = 0; e < 8; ++e) {
    int ent  = t * 8 + e;            // 0..1023
    int l2   = ent & 31;
    int tile = (ent >> 5) & 7;
    int ks   = ent >> 8;
    int h    = tile * 16 + (l2 & 15);
    int kb   = ks * 32 + ((l2 >> 4) << 3);
    const float* src = Wk + h * D_ + kb;
    unsigned short* dst = lds_wk + ent * 16;
    #pragma unroll
    for (int j = 0; j < 16; ++j) dst[j] = f2bf(src[(j & 7) + ((j >> 3) << 4)]);
  }
  lds_hq[t] = hquery[b * D_ + t];
  lds_wa[t] = Wa[t];

  const size_t keyBase = ((size_t)b * N_ + (size_t)ch * CHUNK) * D_;

  // ---- kick off DMA of tile 0 into buffer 0 ----
#if USE_TDM
  if (wv == 0)
    tdm_issue_tile((unsigned)(uintptr_t)&lds_keyf[0][0], key + keyBase, NODES_IT);
#else
  #pragma unroll
  for (int i = 0; i < 16; ++i) {
    int flat = i * THREADS_A + t, row = flat >> 5, c4 = (flat & 31) << 2;
    *(f32x4*)(&lds_keyf[0][row * D_ + c4]) =
        *(const f32x4*)(key + keyBase + (size_t)row * D_ + c4);
  }
#endif
  __syncthreads();

  float hqv[8], wav[8];
  #pragma unroll
  for (int tl = 0; tl < 8; ++tl) {
    hqv[tl] = lds_hq[tl * 16 + (ln & 15)];
    wav[tl] = lds_wa[tl * 16 + (ln & 15)];
  }

  float m_run = -3.0e38f, l_run = 0.0f, acc_k = 0.0f;   // thread t owns k = t

  for (int it = 0; it < ITERS; ++it) {
    const int cur = it & 1;
#if USE_TDM
    if (wv == 0) tdm_wait_all();        // tile `it` resident in LDS
#endif
    __syncthreads();                    // publish tile + retire prev-tile readers

    // ---- issue / perform load of tile it+1 into the other buffer ----
    if (it + 1 < ITERS) {
      const float* nsrc = key + keyBase + (size_t)(it + 1) * NODES_IT * D_;
#if USE_TDM
      if (wv == 0)
        tdm_issue_tile((unsigned)(uintptr_t)&lds_keyf[cur ^ 1][0], nsrc, NODES_IT);
#else
      #pragma unroll
      for (int i = 0; i < 16; ++i) {
        int flat = i * THREADS_A + t, row = flat >> 5, c4 = (flat & 31) << 2;
        *(f32x4*)(&lds_keyf[cur ^ 1][row * D_ + c4]) =
            *(const f32x4*)(nsrc + (size_t)row * D_ + c4);
      }
#endif
    }

    // ---- build all 4 A-fragments first (keeps cvt VALU out of WMMA stream) --
    const float* krow = &lds_keyf[cur][(wv * 16 + (ln & 15)) * D_];
    const int abase = (ln >= 16) ? 8 : 0;
    FragB16 afr[4];
    #pragma unroll
    for (int ks = 0; ks < 4; ++ks) {
      f32x4 c0 = *(const f32x4*)(krow + ks * 32 + abase);
      f32x4 c1 = *(const f32x4*)(krow + ks * 32 + abase + 4);
      f32x4 c2 = *(const f32x4*)(krow + ks * 32 + abase + 16);
      f32x4 c3 = *(const f32x4*)(krow + ks * 32 + abase + 20);
      afr[ks].q[0].x = pk2(c0.x, c0.y); afr[ks].q[0].y = pk2(c0.z, c0.w);
      afr[ks].q[0].z = pk2(c1.x, c1.y); afr[ks].q[0].w = pk2(c1.z, c1.w);
      afr[ks].q[1].x = pk2(c2.x, c2.y); afr[ks].q[1].y = pk2(c2.z, c2.w);
      afr[ks].q[1].z = pk2(c3.x, c3.y); afr[ks].q[1].w = pk2(c3.z, c3.w);
    }

    // ---- per-wave GEMM: 16 nodes x 128 H, K = 128 via 32 bf16 WMMAs ----
    v8f acc[8];
    #pragma unroll
    for (int tl = 0; tl < 8; ++tl) acc[tl] = (v8f)0.0f;
    #pragma unroll
    for (int ks = 0; ks < 4; ++ks) {
      #pragma unroll
      for (int tl = 0; tl < 8; ++tl) {
        FragB16 bf;
        const unsigned short* bent = lds_wk + (((ks * 8 + tl) * 32) + ln) * 16;
        bf.q[0] = *(const u32x4*)(bent);
        bf.q[1] = *(const u32x4*)(bent + 8);
        acc[tl] = __builtin_amdgcn_wmma_f32_16x16x32_bf16(
            false, afr[ks].v, false, bf.v, (short)0, acc[tl], false, false);
      }
    }

    // ---- logits: Wa . leaky_relu(h_key + h_query), reduce over H ----
    #pragma unroll
    for (int r = 0; r < 8; ++r) {
      float s = 0.0f;
      #pragma unroll
      for (int tl = 0; tl < 8; ++tl) {
        float x = acc[tl][r] + hqv[tl];
        x = (x > 0.0f) ? x : NEG_SLOPE * x;
        s += x * wav[tl];
      }
      s += __shfl_xor(s, 1); s += __shfl_xor(s, 2);
      s += __shfl_xor(s, 4); s += __shfl_xor(s, 8);
      if ((ln & 15) == 0) lds_lg[wv * 16 + r + ((ln >> 4) << 3)] = s;
    }
    __syncthreads();

    float lg = 0.0f;
    if (t < NODES_IT) {
      lg = lds_lg[t];
      elog[(size_t)b * N_ + (size_t)ch * CHUNK + it * NODES_IT + t] = lg;
    }

    // ---- online softmax: tile max, weights, rescale, accumulate ----
    if (t < 32) {
      float v = fmaxf(lds_lg[t], lds_lg[t + 32]);
      v = fmaxf(v, __shfl_xor(v, 1));  v = fmaxf(v, __shfl_xor(v, 2));
      v = fmaxf(v, __shfl_xor(v, 4));  v = fmaxf(v, __shfl_xor(v, 8));
      v = fmaxf(v, __shfl_xor(v, 16));
      if (t == 0) lds_red[0] = v;
    }
    __syncthreads();
    float m_new = fmaxf(m_run, lds_red[0]);
    float resc  = expf(m_run - m_new);
    if (t < NODES_IT) lds_w[t] = expf(lg - m_new);
    __syncthreads();
    if (t < 32) {
      float v = lds_w[t] + lds_w[t + 32];
      v += __shfl_xor(v, 1);  v += __shfl_xor(v, 2);
      v += __shfl_xor(v, 4);  v += __shfl_xor(v, 8);
      v += __shfl_xor(v, 16);
      if (t == 0) lds_red[1] = v;
    }
    __syncthreads();
    l_run = l_run * resc + lds_red[1];
    acc_k *= resc;
    #pragma unroll 8
    for (int n = 0; n < NODES_IT; ++n)
      acc_k += lds_w[n] * lds_keyf[cur][n * D_ + t];
    m_run = m_new;
    __syncthreads();
  }

  float* dst = part + (size_t)(b * CHUNKS + ch) * PSTRIDE;
  dst[2 + t] = acc_k;
  if (t == 0) { dst[0] = m_run; dst[1] = l_run; }
}

// ---------------------------------------------------------------------------
// Kernel B: merge chunk partials; scored = Wk @ (ACC/L); elu -> h_prime
// ---------------------------------------------------------------------------
__global__ __launch_bounds__(128)
void ga_combine(const float* __restrict__ part, const float* __restrict__ Wk,
                float* __restrict__ ML, float* __restrict__ outh) {
  __shared__ float sm[CHUNKS], sl[CHUNKS], ss[D_];
  int b = blockIdx.x, t = threadIdx.x;
  if (t < CHUNKS) {
    const float* p = part + (size_t)(b * CHUNKS + t) * PSTRIDE;
    sm[t] = p[0]; sl[t] = p[1];
  }
  __syncthreads();
  float M = -3.0e38f;
  for (int c = 0; c < CHUNKS; ++c) M = fmaxf(M, sm[c]);
  float L = 0.0f, A = 0.0f;
  for (int c = 0; c < CHUNKS; ++c) {
    float f = expf(sm[c] - M);
    L += sl[c] * f;
    A += part[(size_t)(b * CHUNKS + c) * PSTRIDE + 2 + t] * f;
  }
  ss[t] = A / L;
  if (t == 0) { ML[b * 2] = M; ML[b * 2 + 1] = L; }
  __syncthreads();
  float sc = 0.0f;
  #pragma unroll 4
  for (int k = 0; k < D_; ++k) sc += Wk[t * D_ + k] * ss[k];
  outh[b * D_ + t] = (sc > 0.0f) ? sc : (expf(sc) - 1.0f);
}

// ---------------------------------------------------------------------------
// Kernel C: e[b,n] = exp(logit - M_b) / L_b   (in place on d_out e-region)
// ---------------------------------------------------------------------------
__global__ __launch_bounds__(256)
void ga_norm(float* __restrict__ e, const float* __restrict__ ML) {
  int idx = blockIdx.x * 256 + threadIdx.x;     // < B*N
  int b = idx >> 14;                            // N = 16384
  float M = ML[b * 2], L = ML[b * 2 + 1];
  e[idx] = expf(e[idx] - M) / L;
}

// ---------------------------------------------------------------------------
extern "C" void kernel_launch(void* const* d_in, const int* in_sizes, int n_in,
                              void* d_out, int out_size, void* d_ws, size_t ws_size,
                              hipStream_t stream) {
  (void)in_sizes; (void)n_in; (void)out_size; (void)ws_size;
  const float* query = (const float*)d_in[0];
  const float* key   = (const float*)d_in[1];
  const float* Wq    = (const float*)d_in[2];
  const float* Wk    = (const float*)d_in[3];
  const float* Wa    = (const float*)d_in[4];

  float* outh = (float*)d_out;                         // h_prime [B,128]
  float* oute = outh + B_ * D_;                        // e       [B,N]

  float* hq   = (float*)d_ws;                          // B*128
  float* part = hq + B_ * D_;                          // B*CHUNKS*PSTRIDE
  float* ML   = part + (size_t)B_ * CHUNKS * PSTRIDE;  // B*2

  ga_hquery<<<dim3(B_), dim3(128), 0, stream>>>(query, Wq, hq);
  ga_main<<<dim3(CHUNKS, B_), dim3(THREADS_A), 0, stream>>>(key, hq, Wk, Wa, oute, part);
  ga_combine<<<dim3(B_), dim3(128), 0, stream>>>(part, Wk, ML, outh);
  ga_norm<<<dim3((B_ * N_) / 256), dim3(256), 0, stream>>>(oute, ML);
}
